// BilateralCorrelation_8942121910822
// MI455X (gfx1250) — matrix operand order repaired and checked by reading, not compiled, not run on hardware
//
#include <hip/hip_runtime.h>
#include <math.h>

// ---------------------------------------------------------------------------
// Bilateral correlation for MI455X (gfx1250, wave32, WMMA).
//
// Shapes (fixed by setup_inputs): N=2, C=96, H=56, W=80, R=81 (9x9 disp grid).
// Strategy:
//   Pass 1: channel-wise L2 normalize f1/f2 -> f16, channel-LAST layout in ws.
//           (every bilinear corner becomes a contiguous 192B vector; each
//            lane's fragment slice is a 16B-aligned v8h load)
//   Pass 2: one wave per pixel. For each 16-displacement tile, lane L builds
//           A row M=L%16 = bw_feat[r,:] and B col N=L%16 = fw_feat[:,r]
//           (same r per lane -> corners/weights/mask computed once per lane).
//           Bilinear blend runs in PACKED f16 (v_pk_fma_f16, no cvt churn),
//           then v_wmma_f32_16x16x32_f16 over 3 k-chunks (C=96). Output is
//           the diagonal of D (lanes 0-7 / 24-31 per the C/D VGPR layout).
// ---------------------------------------------------------------------------

typedef __attribute__((ext_vector_type(16))) _Float16 v16h;
typedef __attribute__((ext_vector_type(8)))  _Float16 v8h;
typedef __attribute__((ext_vector_type(8)))  float    v8f;

#define N_  2
#define C_  96
#define H_  56
#define W_  80
#define R_  81
#define HW_ (H_ * W_)
#define NP_ (N_ * HW_)   // 8960 pixels

// ---------------- Pass 1: normalize + repack to channel-last f16 -----------
__global__ __launch_bounds__(256)
void normalize_kernel(const float* __restrict__ f1,
                      const float* __restrict__ f2,
                      _Float16* __restrict__ f1n,
                      _Float16* __restrict__ f2n) {
  int p = blockIdx.x * blockDim.x + threadIdx.x;
  if (p >= NP_) return;
  int n = p / HW_, rem = p % HW_;
  const float* b1 = f1 + (size_t)n * C_ * HW_ + rem;
  const float* b2 = f2 + (size_t)n * C_ * HW_ + rem;
  float s1 = 1e-6f, s2 = 1e-6f;                 // eps added to sum-of-squares
  #pragma unroll 8
  for (int c = 0; c < C_; ++c) {
    float a = b1[(size_t)c * HW_]; s1 += a * a;
    float b = b2[(size_t)c * HW_]; s2 += b * b;
  }
  float i1 = 1.0f / sqrtf(s1);
  float i2 = 1.0f / sqrtf(s2);
  _Float16* o1 = f1n + (size_t)p * C_;
  _Float16* o2 = f2n + (size_t)p * C_;
  #pragma unroll 8
  for (int c = 0; c < C_; ++c) {
    o1[c] = (_Float16)(b1[(size_t)c * HW_] * i1);
    o2[c] = (_Float16)(b2[(size_t)c * HW_] * i2);
  }
}

// Per-lane bilinear sample descriptor (border clamp + coverage mask).
struct Samp {
  const _Float16* p00; const _Float16* p01;
  const _Float16* p10; const _Float16* p11;
  float w00, w01, w10, w11;   // wy*wx corner weights
  float mask;                 // cov >= 0.999 ? 1 : 0
};

__device__ inline Samp make_samp(const _Float16* base, float ix, float iy) {
  Samp s;
  float x0f = floorf(ix), y0f = floorf(iy);
  float wx1 = ix - x0f, wx0 = 1.0f - wx1;
  float wy1 = iy - y0f, wy0 = 1.0f - wy1;
  float x1f = x0f + 1.0f, y1f = y0f + 1.0f;
  float vx0 = (x0f >= 0.0f && x0f <= (float)(W_ - 1)) ? 1.0f : 0.0f;
  float vx1 = (x1f >= 0.0f && x1f <= (float)(W_ - 1)) ? 1.0f : 0.0f;
  float vy0 = (y0f >= 0.0f && y0f <= (float)(H_ - 1)) ? 1.0f : 0.0f;
  float vy1 = (y1f >= 0.0f && y1f <= (float)(H_ - 1)) ? 1.0f : 0.0f;
  float cov = wy0 * wx0 * vy0 * vx0 + wy0 * wx1 * vy0 * vx1
            + wy1 * wx0 * vy1 * vx0 + wy1 * wx1 * vy1 * vx1;
  s.mask = (cov >= 0.999f) ? 1.0f : 0.0f;
  int x0c = (int)fminf(fmaxf(x0f, 0.0f), (float)(W_ - 1));
  int x1c = (int)fminf(fmaxf(x1f, 0.0f), (float)(W_ - 1));
  int y0c = (int)fminf(fmaxf(y0f, 0.0f), (float)(H_ - 1));
  int y1c = (int)fminf(fmaxf(y1f, 0.0f), (float)(H_ - 1));
  s.p00 = base + ((size_t)y0c * W_ + x0c) * C_;
  s.p01 = base + ((size_t)y0c * W_ + x1c) * C_;
  s.p10 = base + ((size_t)y1c * W_ + x0c) * C_;
  s.p11 = base + ((size_t)y1c * W_ + x1c) * C_;
  s.w00 = wy0 * wx0; s.w01 = wy0 * wx1;
  s.w10 = wy1 * wx0; s.w11 = wy1 * wx1;
  return s;
}

// ---------------- Pass 2: WMMA correlation ---------------------------------
__global__ __launch_bounds__(256)
void corr_kernel(const float* __restrict__ SBM,
                 const _Float16* __restrict__ f1n,
                 const _Float16* __restrict__ f2n,
                 float* __restrict__ out) {
  const int lane = threadIdx.x & 31;
  const int wave = threadIdx.x >> 5;
  const int p = blockIdx.x * 8 + wave;          // one wave == one pixel
  const int n = p / HW_, rem = p % HW_;
  const int y = rem / W_, x = rem % W_;

  const float sx = SBM[(size_t)(n * 2 + 0) * HW_ + rem];
  const float sy = SBM[(size_t)(n * 2 + 1) * HW_ + rem];

  const _Float16* base1 = f1n + (size_t)n * HW_ * C_;
  const _Float16* base2 = f2n + (size_t)n * HW_ * C_;

  const int half = lane >> 4;     // 0: lanes 0-15, 1: lanes 16-31
  const int m    = lane & 15;     // fragment row (A) == fragment col (B)

  for (int tb = 0; tb < 96; tb += 16) {         // 6 tiles of 16 displacements
    int r = tb + m;
    if (r > 80) r = 80;                         // pad rows duplicate r=80
    const int du = (r % 9) - 4;
    const int dv = (r / 9) - 4;

    // bw: grid - (SBM + d) samples f1 ; fw: grid + (SBM + d) samples f2
    Samp sb = make_samp(base1, (float)x - (sx + (float)du),
                               (float)y - (sy + (float)dv));
    Samp sf = make_samp(base2, (float)x + (sx + (float)du),
                               (float)y + (sy + (float)dv));

    // Hoisted f16 blend weights (broadcast into v_pk_fma_f16 operands).
    const _Float16 hb00 = (_Float16)sb.w00, hb01 = (_Float16)sb.w01;
    const _Float16 hb10 = (_Float16)sb.w10, hb11 = (_Float16)sb.w11;
    const _Float16 hf00 = (_Float16)sf.w00, hf01 = (_Float16)sf.w01;
    const _Float16 hf10 = (_Float16)sf.w10, hf11 = (_Float16)sf.w11;

    v8f acc = {};
    #pragma unroll
    for (int ch = 0; ch < 96; ch += 32) {
      // A (16x32 f16): lane-half K-split {0-7,16-23} / {8-15,24-31}
      const int cA = ch + (half ? 8 : 0);
      // B (32x16 f16): lane-half K-split {0-7,16-23} / {8-15,24-31}
      const int cB = ch + (half ? 8 : 0);

      v8h a00a = *(const v8h*)(sb.p00 + cA); v8h a00b = *(const v8h*)(sb.p00 + cA + 16);
      v8h a01a = *(const v8h*)(sb.p01 + cA); v8h a01b = *(const v8h*)(sb.p01 + cA + 16);
      v8h a10a = *(const v8h*)(sb.p10 + cA); v8h a10b = *(const v8h*)(sb.p10 + cA + 16);
      v8h a11a = *(const v8h*)(sb.p11 + cA); v8h a11b = *(const v8h*)(sb.p11 + cA + 16);

      v8h b00a = *(const v8h*)(sf.p00 + cB); v8h b00b = *(const v8h*)(sf.p00 + cB + 16);
      v8h b01a = *(const v8h*)(sf.p01 + cB); v8h b01b = *(const v8h*)(sf.p01 + cB + 16);
      v8h b10a = *(const v8h*)(sf.p10 + cB); v8h b10b = *(const v8h*)(sf.p10 + cB + 16);
      v8h b11a = *(const v8h*)(sf.p11 + cB); v8h b11b = *(const v8h*)(sf.p11 + cB + 16);

      // Bilinear blend fully in packed f16 (lowers to v_pk_fma_f16).
      v8h vaA = hb00 * a00a + hb01 * a01a + hb10 * a10a + hb11 * a11a;
      v8h vaB = hb00 * a00b + hb01 * a01b + hb10 * a10b + hb11 * a11b;
      v8h vbA = hf00 * b00a + hf01 * b01a + hf10 * b10a + hf11 * b11a;
      v8h vbB = hf00 * b00b + hf01 * b01b + hf10 * b10b + hf11 * b11b;

      v16h afrag = __builtin_shufflevector(vaA, vaB,
          0, 1, 2, 3, 4, 5, 6, 7, 8, 9, 10, 11, 12, 13, 14, 15);
      v16h bfrag = __builtin_shufflevector(vbA, vbB,
          0, 1, 2, 3, 4, 5, 6, 7, 8, 9, 10, 11, 12, 13, 14, 15);

      // D = A(bw) * B(fw) + D ; diagonal = per-displacement channel dot
      acc = __builtin_amdgcn_wmma_f32_16x16x32_f16(
          false, afrag, false, bfrag, (short)0, acc, false, false);
    }

    // Diagonal extraction per C/D layout: diag r lives in lane r (VGPR r) for
    // r<8, and lane r+16 (VGPR r-8) for r>=8  ->  selector k = lane & 7.
    const int k = lane & 7;
    float s01 = (k & 1) ? acc[1] : acc[0];
    float s23 = (k & 1) ? acc[3] : acc[2];
    float s45 = (k & 1) ? acc[5] : acc[4];
    float s67 = (k & 1) ? acc[7] : acc[6];
    float s03 = (k & 2) ? s23 : s01;
    float s47 = (k & 2) ? s67 : s45;
    float dsel = (k & 4) ? s47 : s03;

    const int ridx = tb + m;
    const bool writer = (lane < 8) || (lane >= 24);
    if (writer && ridx < R_) {
      out[(size_t)(n * R_ + ridx) * HW_ + rem] = sb.mask * sf.mask * dsel;
    }
  }
}

// ---------------------------------------------------------------------------
extern "C" void kernel_launch(void* const* d_in, const int* in_sizes, int n_in,
                              void* d_out, int out_size, void* d_ws, size_t ws_size,
                              hipStream_t stream) {
  const float* f1  = (const float*)d_in[0];
  const float* f2  = (const float*)d_in[1];
  const float* SBM = (const float*)d_in[2];

  _Float16* f1n = (_Float16*)d_ws;                 // NP_*96 f16  (1.72 MB)
  _Float16* f2n = f1n + (size_t)NP_ * C_;          // NP_*96 f16  (1.72 MB)

  normalize_kernel<<<(NP_ + 255) / 256, 256, 0, stream>>>(f1, f2, f1n, f2n);
  corr_kernel<<<NP_ / 8, 256, 0, stream>>>(SBM, f1n, f2n, (float*)d_out);
}
